// GPT2_1683627180855
// MI455X (gfx1250) — compile-verified, matrix operand order checked by reading, not run on hardware
//
#include <hip/hip_runtime.h>
#include <hip/hip_bf16.h>

// GPT-2 forward on gfx1250 (MI455X).
// - All GEMMs via V_WMMA_F32_16X16X32_F16 (f16 operands, f32 accumulate).
// - Weights converted fp32->f16 once per launch (halves HBM weight traffic:
//   ~300MB @ 23.3TB/s ~= 13us, vs ~545 GFLOP of WMMA work -> compute-leaning).
// - Double-buffered LDS (48KB), KT=64: one barrier per K-tile, 8 WMMA + 16
//   ds_load_b128 per barrier.
// - Tiles staged with global_load_async_to_lds_b128 (ASYNCcnt path, confirmed
//   declared by this toolchain); fallback register-prefetch pipeline otherwise.
// wave32 fragment layout per CDNA5 ISA 7.12.2: row = lane&15, halves
// K[8*hi..+7] and K[16+8*hi..+7]; C/D: M = vgpr + 8*hi, N = lane&15.

typedef __attribute__((ext_vector_type(16))) _Float16 v16h;
typedef __attribute__((ext_vector_type(8)))  _Float16 v8h;
typedef __attribute__((ext_vector_type(4)))  _Float16 v4h;
typedef __attribute__((ext_vector_type(8)))  float    v8f;
typedef __attribute__((ext_vector_type(4)))  int      v4i;

#define BM 128
#define BN 64
#define KT 64

#define GB   2
#define GS   1024
#define GD   768
#define GH   12
#define GDH  64
#define GF   9216
#define GL   6
#define GV   32000
#define NTOK (GB*GS)

#if defined(__has_builtin)
#if __has_builtin(__builtin_amdgcn_global_load_async_to_lds_b128) && \
    __has_builtin(__builtin_amdgcn_s_wait_asynccnt)
#define USE_ASYNC_LDS 1
#endif
#endif

#ifdef USE_ASYNC_LDS
// signature (from hipcc diagnostic): param0 = v4i addrspace(1)* (non-const),
// param1 = LDS dest, then imm offset + cpol.
#define ASYNC_CP_B128(gsrc, ldst)                                         \
  __builtin_amdgcn_global_load_async_to_lds_b128(                         \
      (__attribute__((address_space(1))) v4i*)(gsrc),                     \
      (__attribute__((address_space(3))) v4i*)(ldst), 0, 0)
#endif

// ---------------- fp32 -> f16 bulk convert (weights) ----------------
__global__ void f32_to_f16_kernel(const float* __restrict__ in, _Float16* __restrict__ out,
                                  long long n4) {
  long long i = (long long)blockIdx.x * blockDim.x + threadIdx.x;
  long long stride = (long long)gridDim.x * blockDim.x;
  for (; i < n4; i += stride) {
    float4 f = ((const float4*)in)[i];
    v4h h; h[0] = (_Float16)f.x; h[1] = (_Float16)f.y; h[2] = (_Float16)f.z; h[3] = (_Float16)f.w;
    ((v4h*)out)[i] = h;
  }
}

// ---------------- embedding gather ----------------
__global__ void embed_kernel(const int* __restrict__ vocab, const int* __restrict__ pos,
                             const float* __restrict__ Ev, const float* __restrict__ Ep,
                             float* __restrict__ x) {
  int t = blockIdx.x;
  int vi = vocab[t], pi = pos[t];
  const float* ev = Ev + (size_t)vi * GD;
  const float* ep = Ep + (size_t)pi * GD;
  float* xo = x + (size_t)t * GD;
  for (int j = threadIdx.x; j < GD; j += 256) xo[j] = ev[j] + ep[j];
}

// ---------------- layernorm (row = 768), f32 and/or f16 outputs ----------------
__global__ void layernorm_kernel(const float* __restrict__ in,
                                 float* __restrict__ outf, _Float16* __restrict__ outh,
                                 const float* __restrict__ g, const float* __restrict__ b) {
  __shared__ float red[256];
  int t = blockIdx.x, tid = threadIdx.x;
  const float* row = in + (size_t)t * GD;
  float s = 0.f;
  for (int j = tid; j < GD; j += 256) s += row[j];
  red[tid] = s; __syncthreads();
  for (int st = 128; st > 0; st >>= 1) { if (tid < st) red[tid] += red[tid + st]; __syncthreads(); }
  float mu = red[0] * (1.0f / (float)GD);
  __syncthreads();
  float v = 0.f;
  for (int j = tid; j < GD; j += 256) { float d = row[j] - mu; v += d * d; }
  red[tid] = v; __syncthreads();
  for (int st = 128; st > 0; st >>= 1) { if (tid < st) red[tid] += red[tid + st]; __syncthreads(); }
  float rstd = rsqrtf(red[0] * (1.0f / (float)GD) + 1e-5f);
  for (int j = tid; j < GD; j += 256) {
    float o = (row[j] - mu) * rstd * g[j] + b[j];
    if (outf) outf[(size_t)t * GD + j] = o;
    if (outh) outh[(size_t)t * GD + j] = (_Float16)o;
  }
}

// ---------------- row softmax (row = 1024), f32 in, f16 out ----------------
__global__ void softmax_kernel(float* __restrict__ p, _Float16* __restrict__ ph) {
  __shared__ float red[256];
  float* row = p + (size_t)blockIdx.x * GS;
  _Float16* oh = ph + (size_t)blockIdx.x * GS;
  int tid = threadIdx.x;
  float m = -3.4e38f;
  for (int j = tid; j < GS; j += 256) m = fmaxf(m, row[j]);
  red[tid] = m; __syncthreads();
  for (int st = 128; st > 0; st >>= 1) { if (tid < st) red[tid] = fmaxf(red[tid], red[tid + st]); __syncthreads(); }
  m = red[0]; __syncthreads();
  float s = 0.f;
  for (int j = tid; j < GS; j += 256) { float e = __expf(row[j] - m); row[j] = e; s += e; }
  red[tid] = s; __syncthreads();
  for (int st = 128; st > 0; st >>= 1) { if (tid < st) red[tid] += red[tid + st]; __syncthreads(); }
  float inv = 1.0f / red[0];
  for (int j = tid; j < GS; j += 256) oh[j] = (_Float16)(row[j] * inv);
}

// ---------------- LDS fragment gather ----------------
__device__ __forceinline__ v16h frag16(const _Float16* base, int hi) {
  v8h lo = *(const v8h*)(base + 8 * hi);
  v8h hq = *(const v8h*)(base + 16 + 8 * hi);
  v16h f;
  #pragma unroll
  for (int q = 0; q < 8; ++q) { f[q] = lo[q]; f[q + 8] = hq[q]; }
  return f;
}

// ---------------- C = act(scale * A·B^T [mask] + bias) [+resid], f32/f16 out ----------------
// A: f16 [M,K] (lda), B: f16 [N,K] (ldb). Batched over z = b*Hdim + h.
__global__ __launch_bounds__(256) void wmma_gemm_nt(
    const _Float16* __restrict__ A, long long lda, long long aStrB, long long aStrH,
    const _Float16* __restrict__ Bm, long long ldb, long long bStrB, long long bStrH,
    float* __restrict__ Cf, _Float16* __restrict__ Ch,
    long long ldc, long long cStrB, long long cStrH,
    const float* __restrict__ bias, const float* __restrict__ resid,
    int K, int Hdim, float scale, int causal, int act)
{
  __shared__ _Float16 As[2][BM * KT];   // 2 x 16 KB
  __shared__ _Float16 Bs[2][BN * KT];   // 2 x  8 KB

  int z = blockIdx.z, bz = z / Hdim, hz = z - bz * Hdim;
  const _Float16* Ap = A  + (size_t)bz * aStrB + (size_t)hz * aStrH;
  const _Float16* Bp = Bm + (size_t)bz * bStrB + (size_t)hz * bStrH;

  int m0 = blockIdx.y * BM, n0 = blockIdx.x * BN;
  int tid = threadIdx.x;
  int lane = tid & 31, wv = tid >> 5;
  int wr = wv & 3, wc = wv >> 2;                 // 4x2 waves over 128x64
  int r = lane & 15, hi = lane >> 4;

  // staging: 16B (8-half) chunks; A: 128 rows x 8 chunks = 1024 -> 4/thread;
  // B: 64 rows x 8 chunks = 512 -> 2/thread.
  int ac = (tid & 7) * 8;
  int arow[4], brow[2];
  const _Float16* ga[4]; const _Float16* gb[2];
  #pragma unroll
  for (int t = 0; t < 4; ++t) {
    arow[t] = (tid + t * 256) >> 3;
    ga[t] = Ap + (size_t)(m0 + arow[t]) * lda + ac;
  }
  #pragma unroll
  for (int t = 0; t < 2; ++t) {
    brow[t] = (tid + t * 256) >> 3;
    gb[t] = Bp + (size_t)(n0 + brow[t]) * ldb + ac;
  }

  v8f acc[2][2] = {};
  const int nT = K / KT;

#ifdef USE_ASYNC_LDS
  // ---- async global->LDS staging (ASYNCcnt) ----
  #pragma unroll
  for (int t = 0; t < 4; ++t) ASYNC_CP_B128(ga[t], &As[0][arow[t] * KT + ac]);
  #pragma unroll
  for (int t = 0; t < 2; ++t) ASYNC_CP_B128(gb[t], &Bs[0][brow[t] * KT + ac]);
  __builtin_amdgcn_s_wait_asynccnt(0);
  __syncthreads();
  for (int it = 0; it < nT; ++it) {
    int buf = it & 1;
    if (it + 1 < nT) {
      long long kn = (long long)(it + 1) * KT;
      #pragma unroll
      for (int t = 0; t < 4; ++t) ASYNC_CP_B128(ga[t] + kn, &As[1 - buf][arow[t] * KT + ac]);
      #pragma unroll
      for (int t = 0; t < 2; ++t) ASYNC_CP_B128(gb[t] + kn, &Bs[1 - buf][brow[t] * KT + ac]);
    }
    #pragma unroll
    for (int ks = 0; ks < 2; ++ks) {
      v16h af[2], bf[2];
      #pragma unroll
      for (int i = 0; i < 2; ++i) af[i] = frag16(&As[buf][(wr * 32 + i * 16 + r) * KT + ks * 32], hi);
      #pragma unroll
      for (int j = 0; j < 2; ++j) bf[j] = frag16(&Bs[buf][(wc * 32 + j * 16 + r) * KT + ks * 32], hi);
      #pragma unroll
      for (int i = 0; i < 2; ++i)
        #pragma unroll
        for (int j = 0; j < 2; ++j)
          acc[i][j] = __builtin_amdgcn_wmma_f32_16x16x32_f16(
              false, af[i], false, bf[j], (short)0, acc[i][j], false, false);
    }
    if (it + 1 < nT) __builtin_amdgcn_s_wait_asynccnt(0);
    __syncthreads();
  }
#else
  // ---- register-prefetch staging ----
  v8h pa[4], pb[2];
  #pragma unroll
  for (int t = 0; t < 4; ++t) pa[t] = *(const v8h*)ga[t];
  #pragma unroll
  for (int t = 0; t < 2; ++t) pb[t] = *(const v8h*)gb[t];
  #pragma unroll
  for (int t = 0; t < 4; ++t) *(v8h*)&As[0][arow[t] * KT + ac] = pa[t];
  #pragma unroll
  for (int t = 0; t < 2; ++t) *(v8h*)&Bs[0][brow[t] * KT + ac] = pb[t];
  __syncthreads();
  for (int it = 0; it < nT; ++it) {
    int buf = it & 1;
    if (it + 1 < nT) {
      long long kn = (long long)(it + 1) * KT;
      #pragma unroll
      for (int t = 0; t < 4; ++t) pa[t] = *(const v8h*)(ga[t] + kn);
      #pragma unroll
      for (int t = 0; t < 2; ++t) pb[t] = *(const v8h*)(gb[t] + kn);
      if (it + 2 < nT) {
        __builtin_prefetch(ga[0] + kn + KT, 0, 1);
        __builtin_prefetch(gb[0] + kn + KT, 0, 1);
      }
    }
    #pragma unroll
    for (int ks = 0; ks < 2; ++ks) {
      v16h af[2], bf[2];
      #pragma unroll
      for (int i = 0; i < 2; ++i) af[i] = frag16(&As[buf][(wr * 32 + i * 16 + r) * KT + ks * 32], hi);
      #pragma unroll
      for (int j = 0; j < 2; ++j) bf[j] = frag16(&Bs[buf][(wc * 32 + j * 16 + r) * KT + ks * 32], hi);
      #pragma unroll
      for (int i = 0; i < 2; ++i)
        #pragma unroll
        for (int j = 0; j < 2; ++j)
          acc[i][j] = __builtin_amdgcn_wmma_f32_16x16x32_f16(
              false, af[i], false, bf[j], (short)0, acc[i][j], false, false);
    }
    if (it + 1 < nT) {
      #pragma unroll
      for (int t = 0; t < 4; ++t) *(v8h*)&As[1 - buf][arow[t] * KT + ac] = pa[t];
      #pragma unroll
      for (int t = 0; t < 2; ++t) *(v8h*)&Bs[1 - buf][brow[t] * KT + ac] = pb[t];
    }
    __syncthreads();
  }
#endif

  float* Cfp = Cf ? Cf + (size_t)bz * cStrB + (size_t)hz * cStrH : nullptr;
  _Float16* Chp = Ch ? Ch + (size_t)bz * cStrB + (size_t)hz * cStrH : nullptr;
  #pragma unroll
  for (int i = 0; i < 2; ++i) {
    #pragma unroll
    for (int j = 0; j < 2; ++j) {
      int nn = n0 + wc * 32 + j * 16 + r;
      float bv = bias ? bias[nn] : 0.0f;
      #pragma unroll
      for (int rr = 0; rr < 8; ++rr) {
        int mm = m0 + wr * 32 + i * 16 + rr + 8 * hi;
        float v = acc[i][j][rr] * scale + bv;
        if (causal && nn > mm) v = -100000.0f;         // w*mask - (1-mask)*1e5
        if (act == 1) v = v > 0.f ? v : 0.01f * v;     // leaky_relu(0.01)
        if (resid) v += resid[(size_t)mm * ldc + nn];
        if (Cfp) Cfp[(size_t)mm * ldc + nn] = v;
        if (Chp) Chp[(size_t)mm * ldc + nn] = (_Float16)v;
      }
    }
  }
}

// ---------------- C(f16) = A·B, B is f16 [K,N] row stride ldb (probs @ V) ----------------
__global__ __launch_bounds__(256) void wmma_gemm_nn(
    const _Float16* __restrict__ A, long long lda, long long aStrB, long long aStrH,
    const _Float16* __restrict__ Bm, long long ldb, long long bStrB, long long bStrH,
    _Float16* __restrict__ C, long long ldc, long long cStrB, long long cStrH,
    int K, int Hdim)
{
  __shared__ _Float16 As[2][BM * KT];
  __shared__ _Float16 Bs[2][BN * KT];   // transposed: Bs[n*KT + kk]

  int z = blockIdx.z, bz = z / Hdim, hz = z - bz * Hdim;
  const _Float16* Ap = A  + (size_t)bz * aStrB + (size_t)hz * aStrH;
  const _Float16* Bp = Bm + (size_t)bz * bStrB + (size_t)hz * bStrH;

  int m0 = blockIdx.y * BM, n0 = blockIdx.x * BN;
  int tid = threadIdx.x;
  int lane = tid & 31, wv = tid >> 5;
  int wr = wv & 3, wc = wv >> 2;
  int r = lane & 15, hi = lane >> 4;

  int ac = (tid & 7) * 8;
  int arow[4];
  const _Float16* ga[4];
  #pragma unroll
  for (int t = 0; t < 4; ++t) {
    arow[t] = (tid + t * 256) >> 3;
    ga[t] = Ap + (size_t)(m0 + arow[t]) * lda + ac;
  }
  // B tile: 64 kk-rows x 64 n; 16B chunk per (kk, 8n); 512 chunks -> 2/thread
  int bkk[2];
  #pragma unroll
  for (int t = 0; t < 2; ++t) bkk[t] = (tid + t * 256) >> 3;
  int bn8 = (tid & 7) * 8;

  v8h pa[4], pb[2];
  #pragma unroll
  for (int t = 0; t < 4; ++t) pa[t] = *(const v8h*)ga[t];
  #pragma unroll
  for (int t = 0; t < 2; ++t) pb[t] = *(const v8h*)(Bp + (size_t)bkk[t] * ldb + n0 + bn8);

  #pragma unroll
  for (int t = 0; t < 4; ++t) *(v8h*)&As[0][arow[t] * KT + ac] = pa[t];
  #pragma unroll
  for (int t = 0; t < 2; ++t)
    #pragma unroll
    for (int q = 0; q < 8; ++q) Bs[0][(bn8 + q) * KT + bkk[t]] = pb[t][q];
  __syncthreads();

  v8f acc[2][2] = {};
  const int nT = K / KT;
  for (int it = 0; it < nT; ++it) {
    int buf = it & 1;
    if (it + 1 < nT) {
      long long kn = (long long)(it + 1) * KT;
      #pragma unroll
      for (int t = 0; t < 4; ++t) pa[t] = *(const v8h*)(ga[t] + kn);
      #pragma unroll
      for (int t = 0; t < 2; ++t) pb[t] = *(const v8h*)(Bp + (size_t)(kn + bkk[t]) * ldb + n0 + bn8);
    }
    #pragma unroll
    for (int ks = 0; ks < 2; ++ks) {
      v16h af[2], bf[2];
      #pragma unroll
      for (int i = 0; i < 2; ++i) af[i] = frag16(&As[buf][(wr * 32 + i * 16 + r) * KT + ks * 32], hi);
      #pragma unroll
      for (int j = 0; j < 2; ++j) bf[j] = frag16(&Bs[buf][(wc * 32 + j * 16 + r) * KT + ks * 32], hi);
      #pragma unroll
      for (int i = 0; i < 2; ++i)
        #pragma unroll
        for (int j = 0; j < 2; ++j)
          acc[i][j] = __builtin_amdgcn_wmma_f32_16x16x32_f16(
              false, af[i], false, bf[j], (short)0, acc[i][j], false, false);
    }
    if (it + 1 < nT) {
      #pragma unroll
      for (int t = 0; t < 4; ++t) *(v8h*)&As[1 - buf][arow[t] * KT + ac] = pa[t];
      #pragma unroll
      for (int t = 0; t < 2; ++t)
        #pragma unroll
        for (int q = 0; q < 8; ++q) Bs[1 - buf][(bn8 + q) * KT + bkk[t]] = pb[t][q];
    }
    __syncthreads();
  }

  _Float16* Cp = C + (size_t)bz * cStrB + (size_t)hz * cStrH;
  #pragma unroll
  for (int i = 0; i < 2; ++i)
    #pragma unroll
    for (int j = 0; j < 2; ++j) {
      int nn = n0 + wc * 32 + j * 16 + r;
      #pragma unroll
      for (int rr = 0; rr < 8; ++rr) {
        int mm = m0 + wr * 32 + i * 16 + rr + 8 * hi;
        Cp[(size_t)mm * ldc + nn] = (_Float16)acc[i][j][rr];
      }
    }
}

extern "C" void kernel_launch(void* const* d_in, const int* in_sizes, int n_in,
                              void* d_out, int out_size, void* d_ws, size_t ws_size,
                              hipStream_t stream) {
  (void)in_sizes; (void)n_in; (void)out_size; (void)ws_size;
  const int*   vocab = (const int*)d_in[0];
  const int*   pos   = (const int*)d_in[1];
  const float* Ev    = (const float*)d_in[2];
  const float* Ep    = (const float*)d_in[3];
  const float* ln1g  = (const float*)d_in[4];
  const float* ln1b  = (const float*)d_in[5];
  const float* Wqkv  = (const float*)d_in[6];
  const float* bqkv  = (const float*)d_in[7];
  const float* Wo    = (const float*)d_in[8];
  const float* bo    = (const float*)d_in[9];
  const float* ln2g  = (const float*)d_in[10];
  const float* ln2b  = (const float*)d_in[11];
  const float* W1    = (const float*)d_in[12];
  const float* b1    = (const float*)d_in[13];
  const float* W2    = (const float*)d_in[14];
  const float* b2    = (const float*)d_in[15];
  const float* Whead = (const float*)d_in[16];
  float* out = (float*)d_out;

  // ---- workspace carve-up: f32 region then f16 region ----
  char* wsb = (char*)d_ws;
  size_t ob = 0;
  float* x       = (float*)(wsb + ob); ob += (size_t)NTOK * GD * 4;
  float* probs   = (float*)(wsb + ob); ob += (size_t)GB * GH * GS * GS * 4;
  float* attnres = (float*)(wsb + ob); ob += (size_t)NTOK * GD * 4;
  float* n2      = (float*)(wsb + ob); ob += (size_t)NTOK * GD * 4;

  _Float16* n1h      = (_Float16*)(wsb + ob); ob += (size_t)NTOK * GD * 2;
  _Float16* qkvh     = (_Float16*)(wsb + ob); ob += (size_t)NTOK * 3 * GD * 2;
  _Float16* probsh   = (_Float16*)(wsb + ob); ob += (size_t)GB * GH * GS * GS * 2;
  _Float16* attnouth = (_Float16*)(wsb + ob); ob += (size_t)NTOK * GD * 2;
  _Float16* n2h      = (_Float16*)(wsb + ob); ob += (size_t)NTOK * GD * 2;
  _Float16* hbufh    = (_Float16*)(wsb + ob); ob += (size_t)NTOK * GF * 2;
  _Float16* xh       = (_Float16*)(wsb + ob); ob += (size_t)NTOK * GD * 2;

  _Float16* Wqkvh  = (_Float16*)(wsb + ob); ob += (size_t)GL * 3 * GD * GD * 2;
  _Float16* Woh    = (_Float16*)(wsb + ob); ob += (size_t)GL * GD * GD * 2;
  _Float16* W1h    = (_Float16*)(wsb + ob); ob += (size_t)GL * GF * GD * 2;
  _Float16* W2h    = (_Float16*)(wsb + ob); ob += (size_t)GL * GD * GF * 2;
  _Float16* Wheadh = (_Float16*)(wsb + ob); ob += (size_t)GV * GD * 2;

  dim3 blk(256);

  // ---- weights fp32 -> f16 (once per launch) ----
  f32_to_f16_kernel<<<4096, 256, 0, stream>>>(Wqkv,  Wqkvh,  (long long)GL * 3 * GD * GD / 4);
  f32_to_f16_kernel<<<4096, 256, 0, stream>>>(Wo,    Woh,    (long long)GL * GD * GD / 4);
  f32_to_f16_kernel<<<4096, 256, 0, stream>>>(W1,    W1h,    (long long)GL * GF * GD / 4);
  f32_to_f16_kernel<<<4096, 256, 0, stream>>>(W2,    W2h,    (long long)GL * GD * GF / 4);
  f32_to_f16_kernel<<<4096, 256, 0, stream>>>(Whead, Wheadh, (long long)GV * GD / 4);

  const long long qkvStrB = (long long)GS * 3 * GD;
  const long long pStrB   = (long long)GH * GS * GS;
  const long long pStrH   = (long long)GS * GS;

  embed_kernel<<<NTOK, 256, 0, stream>>>(vocab, pos, Ev, Ep, x);

  for (int i = 0; i < GL; ++i) {
    layernorm_kernel<<<NTOK, 256, 0, stream>>>(x, nullptr, n1h,
        ln1g + (size_t)i * GD, ln1b + (size_t)i * GD);

    // qkv(f16) = n1 @ Wqkv^T + bqkv     [2048 x 2304]
    wmma_gemm_nt<<<dim3(3 * GD / BN, NTOK / BM, 1), blk, 0, stream>>>(
        n1h, GD, 0, 0,
        Wqkvh + (size_t)i * 3 * GD * GD, GD, 0, 0,
        nullptr, qkvh, 3 * GD, 0, 0,
        bqkv + (size_t)i * 3 * GD, nullptr,
        GD, 1, 1.0f, 0, 0);

    // scores(f32) = (q @ k^T)/8, causal masked   [B,H,S,S]
    wmma_gemm_nt<<<dim3(GS / BN, GS / BM, GB * GH), blk, 0, stream>>>(
        qkvh,       3 * GD, qkvStrB, 3 * GDH,
        qkvh + GDH, 3 * GD, qkvStrB, 3 * GDH,
        probs, nullptr, GS, pStrB, pStrH,
        nullptr, nullptr,
        GDH, GH, 0.125f, 1, 0);

    softmax_kernel<<<GB * GH * GS, 256, 0, stream>>>(probs, probsh);

    // attnout(f16)[b,s,h*64+d] = probs @ V
    wmma_gemm_nn<<<dim3(GDH / BN, GS / BM, GB * GH), blk, 0, stream>>>(
        probsh, GS, pStrB, pStrH,
        qkvh + 2 * GDH, 3 * GD, qkvStrB, 3 * GDH,
        attnouth, GD, (long long)GS * GD, GDH,
        GS, GH);

    // attnres(f32) = attnout @ Wo^T + bo + x
    wmma_gemm_nt<<<dim3(GD / BN, NTOK / BM, 1), blk, 0, stream>>>(
        attnouth, GD, 0, 0,
        Woh + (size_t)i * GD * GD, GD, 0, 0,
        attnres, nullptr, GD, 0, 0,
        bo + (size_t)i * GD, x,
        GD, 1, 1.0f, 0, 0);

    layernorm_kernel<<<NTOK, 256, 0, stream>>>(attnres, n2, n2h,
        ln2g + (size_t)i * GD, ln2b + (size_t)i * GD);

    // h(f16) = leaky_relu(n2 @ W1^T + b1)   [2048 x 9216]
    wmma_gemm_nt<<<dim3(GF / BN, NTOK / BM, 1), blk, 0, stream>>>(
        n2h, GD, 0, 0,
        W1h + (size_t)i * GF * GD, GD, 0, 0,
        nullptr, hbufh, GF, 0, 0,
        b1 + (size_t)i * GF, nullptr,
        GD, 1, 1.0f, 0, 1);

    // x(f32)+xh(f16) = h @ W2^T + b2 + n2
    wmma_gemm_nt<<<dim3(GD / BN, NTOK / BM, 1), blk, 0, stream>>>(
        hbufh, GF, 0, 0,
        W2h + (size_t)i * GD * GF, GF, 0, 0,
        x, xh, GD, 0, 0,
        b2 + (size_t)i * GD, n2,
        GF, 1, 1.0f, 0, 0);
  }

  // logits(f32) = x @ W_head^T    [2048 x 32000]
  wmma_gemm_nt<<<dim3(GV / BN, NTOK / BM, 1), blk, 0, stream>>>(
      xh, GD, 0, 0,
      Wheadh, GD, 0, 0,
      out, nullptr, GV, 0, 0,
      nullptr, nullptr,
      GD, 1, 1.0f, 0, 0);
}
